// GAT_51007031608158
// MI455X (gfx1250) — matrix operand order repaired and checked by reading, not compiled, not run on hardware
//
#include <hip/hip_runtime.h>
#include <hip/hip_bf16.h>
#include <math.h>

// ---------------- types ----------------
typedef __attribute__((ext_vector_type(16))) __bf16          v16bf;
typedef __attribute__((ext_vector_type(8)))  float           v8f;
typedef __attribute__((ext_vector_type(8)))  unsigned short  ush8;

#define F_IN 128
#define H1   64
#define NEG_SLOPE 0.2f
#define BN_EPS    1e-5f

// ---------------- helpers ----------------
__device__ __forceinline__ unsigned short f2bf(float f) {
  unsigned u = __float_as_uint(f);
  u += 0x7FFFu + ((u >> 16) & 1u);          // round to nearest even
  return (unsigned short)(u >> 16);
}
// order-preserving float<->u32 map for atomicMax on floats
__device__ __forceinline__ unsigned f2ord(float f) {
  unsigned u = __float_as_uint(f);
  return (u & 0x80000000u) ? ~u : (u | 0x80000000u);
}
__device__ __forceinline__ float ord2f(unsigned u) {
  return (u & 0x80000000u) ? __uint_as_float(u & 0x7FFFFFFFu) : __uint_as_float(~u);
}
__device__ __forceinline__ void edge_sd(const long long* __restrict__ src,
                                        const long long* __restrict__ dst,
                                        int E, int i, int& s, int& d) {
  if (i < E) { s = (int)src[i]; d = (int)dst[i]; }
  else       { s = i - E; d = i - E; }        // appended self-loops
}

// ---------------- prep kernels ----------------
__global__ void k_cast_bf16(const float* __restrict__ x, unsigned short* __restrict__ xb,
                            long long n) {
  long long i = (long long)blockIdx.x * blockDim.x + threadIdx.x;
  if (i < n) xb[i] = f2bf(x[i]);
}

// Pack W1_src (128x64 f32, row-major K x Ncol) into per-lane WMMA-B fragments:
// packed[((kt*4+ct)*32 + lane)*16 + j] = bf16(W[(kt*32 + (lane>>4)*8 + (j&7) + (j>>3)*16)*64 + ct*16 + (lane&15)])
__global__ void k_pack_w1(const float* __restrict__ W, unsigned short* __restrict__ Wp) {
  int idx = blockIdx.x * blockDim.x + threadIdx.x;
  if (idx >= 16 * 32 * 16) return;
  int j    = idx & 15;
  int l    = (idx >> 4) & 31;
  int tile = idx >> 9;                 // kt*4 + ct
  int kt = tile >> 2, ct = tile & 3;
  int K   = kt * 32 + (l >> 4) * 8 + (j & 7) + (j >> 3) * 16;
  int col = ct * 16 + (l & 15);
  Wp[idx] = f2bf(W[(size_t)K * H1 + col]);
}

// wd[k] = sum_j Wd[k*C + j] * a[j]   (folds W_dst @ a_dst)
__global__ void k_wdvec(const float* __restrict__ Wd, const float* __restrict__ a,
                        float* __restrict__ wd, int K, int C) {
  int k = blockIdx.x * blockDim.x + threadIdx.x;
  if (k >= K) return;
  float s = 0.f;
  for (int j = 0; j < C; ++j) s += Wd[(size_t)k * C + j] * a[j];
  wd[k] = s;
}

// ---------------- layer-1 GEMM: hs = x @ W1_src via bf16 WMMA ----------------
// One wave computes a 16x64 output tile: 4 k-steps (K=32 each) x 4 column tiles.
__global__ void __launch_bounds__(256)
k_gemm1_wmma(const unsigned short* __restrict__ xb, const unsigned short* __restrict__ Wp,
             float* __restrict__ hs, int N) {
  int gtid = blockIdx.x * blockDim.x + threadIdx.x;
  int wave = gtid >> 5;
  int lane = gtid & 31;
  int row0 = wave * 16;
  if (row0 >= N) return;                      // wave-uniform exit (EXEC stays all-ones)
  int lh = lane >> 4;                         // lane half: K offset group
  int lm = lane & 15;                         // matrix row (A) / column (B,C)
  int ar = row0 + lm; if (ar >= N) ar = N - 1;
  const unsigned short* arow = xb + (size_t)ar * F_IN;

  v8f acc0 = {}, acc1 = {}, acc2 = {}, acc3 = {};
  union { ush8 h[2]; v16bf v; } A;
#pragma unroll
  for (int kt = 0; kt < 4; ++kt) {
    int kb = kt * 32 + lh * 8;                // 16-bit A layout: K = kb..kb+7, kb+16..kb+23
    A.h[0] = *(const ush8*)(arow + kb);
    A.h[1] = *(const ush8*)(arow + kb + 16);
    const unsigned short* wp = Wp + (size_t)(kt * 4) * 512 + (size_t)lane * 16;
    v16bf b0 = *(const v16bf*)(wp + 0 * 512);
    v16bf b1 = *(const v16bf*)(wp + 1 * 512);
    v16bf b2 = *(const v16bf*)(wp + 2 * 512);
    v16bf b3 = *(const v16bf*)(wp + 3 * 512);
    acc0 = __builtin_amdgcn_wmma_f32_16x16x32_bf16(false, A.v, false, b0, (short)0, acc0, false, false);
    acc1 = __builtin_amdgcn_wmma_f32_16x16x32_bf16(false, A.v, false, b1, (short)0, acc1, false, false);
    acc2 = __builtin_amdgcn_wmma_f32_16x16x32_bf16(false, A.v, false, b2, (short)0, acc2, false, false);
    acc3 = __builtin_amdgcn_wmma_f32_16x16x32_bf16(false, A.v, false, b3, (short)0, acc3, false, false);
  }
  // C/D layout: lane lm = column, VGPR i = row (i + lh*8)
  int rbase = row0 + lh * 8;
#pragma unroll
  for (int i = 0; i < 8; ++i) {
    int r = rbase + i;
    if (r < N) {
      float* orow = hs + (size_t)r * H1 + lm;
      orow[0]  = acc0[i];
      orow[16] = acc1[i];
      orow[32] = acc2[i];
      orow[48] = acc3[i];
    }
  }
}

// alpha_s[n] = hs[n] . a1_src ; alpha_d[n] = x[n] . wd1
__global__ void k_alpha1(const float* __restrict__ hs, const float* __restrict__ x,
                         const float* __restrict__ a1s, const float* __restrict__ wd1,
                         float* __restrict__ as1, float* __restrict__ ad1, int N) {
  int n = blockIdx.x * blockDim.x + threadIdx.x;
  if (n >= N) return;
  const float* hr = hs + (size_t)n * H1;
  float a = 0.f;
  for (int j = 0; j < H1; ++j) a += hr[j] * a1s[j];
  const float* xr = x + (size_t)n * F_IN;
  float d = 0.f;
  for (int k = 0; k < F_IN; ++k) d += xr[k] * wd1[k];
  as1[n] = a; ad1[n] = d;
}

// ---------------- fills ----------------
__global__ void k_fill_f32(float* __restrict__ p, float v, long long n) {
  long long i = (long long)blockIdx.x * blockDim.x + threadIdx.x;
  if (i < n) p[i] = v;
}
__global__ void k_fill_u32(unsigned* __restrict__ p, unsigned v, long long n) {
  long long i = (long long)blockIdx.x * blockDim.x + threadIdx.x;
  if (i < n) p[i] = v;
}
__global__ void k_init_out2(float* __restrict__ out, const float* __restrict__ b2, int N) {
  int i = blockIdx.x * blockDim.x + threadIdx.x;
  if (i < 2 * N) out[i] = b2[i & 1];
}

// ---------------- edge passes (shared by both layers) ----------------
__global__ void k_edge_score(const long long* __restrict__ src, const long long* __restrict__ dst,
                             int E, int N, const float* __restrict__ as, const float* __restrict__ ad,
                             float* __restrict__ e, unsigned* __restrict__ m) {
  int i = blockIdx.x * blockDim.x + threadIdx.x;
  if (i >= E + N) return;
  int s, d; edge_sd(src, dst, E, i, s, d);
  float v = as[s] + ad[d];
  v = (v > 0.f) ? v : NEG_SLOPE * v;          // leaky_relu
  e[i] = v;
  atomicMax(&m[d], f2ord(v));                 // segment max
}

__global__ void k_edge_expsum(const long long* __restrict__ src, const long long* __restrict__ dst,
                              int E, int N, float* __restrict__ e, const unsigned* __restrict__ m,
                              float* __restrict__ sden) {
  int i = blockIdx.x * blockDim.x + threadIdx.x;
  if (i >= E + N) return;
  int s, d; edge_sd(src, dst, E, i, s, d);
  float p = expf(e[i] - ord2f(m[d]));
  e[i] = p;                                   // overwrite e with p
  atomicAdd(&sden[d], p);                     // segment sum
}

// layer-1 scatter: 8 threads per edge, 8 channels each
__global__ void k_edge_aggr64(const long long* __restrict__ src, const long long* __restrict__ dst,
                              int E, int N, const float* __restrict__ p, const float* __restrict__ sden,
                              const float* __restrict__ hs, float* __restrict__ out) {
  long long t = (long long)blockIdx.x * blockDim.x + threadIdx.x;
  long long total = (long long)(E + N) * 8;
  if (t >= total) return;
  int i  = (int)(t >> 3);
  int cg = (int)(t & 7) * 8;
  int s, d; edge_sd(src, dst, E, i, s, d);
  float alpha = p[i] / (sden[d] + 1e-16f);
  const float* hr = hs + (size_t)s * H1 + cg;
  float* orow = out + (size_t)d * H1 + cg;
#pragma unroll
  for (int c = 0; c < 8; ++c) atomicAdd(&orow[c], hr[c] * alpha);
}

// layer-2 scatter: 2 channels per edge
__global__ void k_edge_aggr2(const long long* __restrict__ src, const long long* __restrict__ dst,
                             int E, int N, const float* __restrict__ p, const float* __restrict__ sden,
                             const float* __restrict__ hs2, float* __restrict__ out) {
  int i = blockIdx.x * blockDim.x + threadIdx.x;
  if (i >= E + N) return;
  int s, d; edge_sd(src, dst, E, i, s, d);
  float alpha = p[i] / (sden[d] + 1e-16f);
  atomicAdd(&out[(size_t)d * 2 + 0], hs2[(size_t)s * 2 + 0] * alpha);
  atomicAdd(&out[(size_t)d * 2 + 1], hs2[(size_t)s * 2 + 1] * alpha);
}

// ---------------- batch norm ----------------
__global__ void __launch_bounds__(256)
k_bn_stats(const float* __restrict__ h, const float* __restrict__ b1,
           float* __restrict__ sum, float* __restrict__ sumsq, int N) {
  __shared__ float ls[256], lq[256];
  int c = threadIdx.x & 63, sub = threadIdx.x >> 6;
  float bb = b1[c];
  float s = 0.f, q = 0.f;
  for (int r = blockIdx.x * 4 + sub; r < N; r += gridDim.x * 4) {
    float v = h[(size_t)r * H1 + c] + bb;
    s += v; q += v * v;
  }
  ls[threadIdx.x] = s; lq[threadIdx.x] = q;
  __syncthreads();
  if (sub == 0) {
    s = ls[c] + ls[c + 64] + ls[c + 128] + ls[c + 192];
    q = lq[c] + lq[c + 64] + lq[c + 128] + lq[c + 192];
    atomicAdd(&sum[c], s);
    atomicAdd(&sumsq[c], q);
  }
}

__global__ void k_bn_final(const float* __restrict__ sum, const float* __restrict__ sumsq,
                           const float* __restrict__ gamma, const float* __restrict__ beta,
                           float* __restrict__ scale, float* __restrict__ shift, int N) {
  int c = threadIdx.x;
  if (c >= H1) return;
  float inv = 1.0f / (float)N;
  float mu  = sum[c] * inv;
  float var = sumsq[c] * inv - mu * mu;
  float sc  = gamma[c] * rsqrtf(var + BN_EPS);
  scale[c] = sc;
  shift[c] = beta[c] - mu * sc;
}

__global__ void k_bn_apply(float* __restrict__ h, const float* __restrict__ b1,
                           const float* __restrict__ scale, const float* __restrict__ shift,
                           long long n) {
  long long i = (long long)blockIdx.x * blockDim.x + threadIdx.x;
  if (i >= n) return;
  int c = (int)(i & 63);
  float v = (h[i] + b1[c]) * scale[c] + shift[c];
  h[i] = v > 0.f ? v : 0.f;                   // relu
}

// ---------------- layer-2 projection (tiny GEMM, scalar) ----------------
__global__ void k_proj2(const float* __restrict__ h, const float* __restrict__ W2s,
                        const float* __restrict__ a2s, const float* __restrict__ wd2,
                        float* __restrict__ hs2, float* __restrict__ as2,
                        float* __restrict__ ad2, int N) {
  int n = blockIdx.x * blockDim.x + threadIdx.x;
  if (n >= N) return;
  const float* hr = h + (size_t)n * H1;
  float c0 = 0.f, c1 = 0.f, dd = 0.f;
  for (int k = 0; k < H1; ++k) {
    float v = hr[k];
    c0 += v * W2s[k * 2 + 0];
    c1 += v * W2s[k * 2 + 1];
    dd += v * wd2[k];
  }
  hs2[(size_t)n * 2 + 0] = c0;
  hs2[(size_t)n * 2 + 1] = c1;
  as2[n] = c0 * a2s[0] + c1 * a2s[1];
  ad2[n] = dd;
}

// ---------------- launch ----------------
extern "C" void kernel_launch(void* const* d_in, const int* in_sizes, int n_in,
                              void* d_out, int out_size, void* d_ws, size_t ws_size,
                              hipStream_t stream) {
  (void)n_in; (void)out_size; (void)ws_size;
  const float*     x   = (const float*)d_in[0];
  const long long* ei  = (const long long*)d_in[1];   // int64 in reference
  const float* W1s = (const float*)d_in[2];
  const float* W1d = (const float*)d_in[3];
  const float* a1s = (const float*)d_in[4];
  const float* a1d = (const float*)d_in[5];
  const float* b1  = (const float*)d_in[6];
  const float* gam = (const float*)d_in[7];
  const float* bet = (const float*)d_in[8];
  const float* W2s = (const float*)d_in[9];
  const float* W2d = (const float*)d_in[10];
  const float* a2s = (const float*)d_in[11];
  const float* a2d = (const float*)d_in[12];
  const float* b2  = (const float*)d_in[13];

  const int N = in_sizes[0] / F_IN;
  const int E = in_sizes[1] / 2;
  const long long* esrc = ei;
  const long long* edst = ei + E;
  const int T = E + N;                         // edges incl. self-loops

  // workspace layout (256B aligned slices)
  char* ws = (char*)d_ws;
  size_t off = 0;
  auto alloc = [&](size_t bytes) -> char* {
    char* p = ws + off;
    off += (bytes + 255) & ~(size_t)255;
    return p;
  };
  unsigned short* xb  = (unsigned short*)alloc((size_t)N * F_IN * 2);
  unsigned short* Wp  = (unsigned short*)alloc(16 * 32 * 16 * 2);
  float*    hs   = (float*)alloc((size_t)N * H1 * 4);
  float*    wd1  = (float*)alloc(F_IN * 4);
  float*    as1  = (float*)alloc((size_t)N * 4);
  float*    ad1  = (float*)alloc((size_t)N * 4);
  float*    e1   = (float*)alloc((size_t)T * 4);
  unsigned* m1   = (unsigned*)alloc((size_t)N * 4);
  float*    s1   = (float*)alloc((size_t)N * 4);
  float*    out1 = (float*)alloc((size_t)N * H1 * 4);
  float*    bnsum = (float*)alloc(128 * 4);    // [0:64) sum, [64:128) sumsq
  float*    bnsq  = bnsum + 64;
  float*    bnscale = (float*)alloc(64 * 4);
  float*    bnshift = (float*)alloc(64 * 4);
  float*    wd2  = (float*)alloc(64 * 4);
  float*    hs2  = (float*)alloc((size_t)N * 2 * 4);
  float*    as2  = (float*)alloc((size_t)N * 4);
  float*    ad2  = (float*)alloc((size_t)N * 4);
  float*    e2   = (float*)alloc((size_t)T * 4);
  unsigned* m2   = (unsigned*)alloc((size_t)N * 4);
  float*    s2   = (float*)alloc((size_t)N * 4);

  const int B = 256;
  auto nb = [](long long n, int b) { return (int)((n + b - 1) / b); };

  // ---- layer 1 dense ----
  k_cast_bf16<<<nb((long long)N * F_IN, B), B, 0, stream>>>(x, xb, (long long)N * F_IN);
  k_pack_w1<<<nb(16 * 32 * 16, B), B, 0, stream>>>(W1s, Wp);
  k_wdvec<<<1, F_IN, 0, stream>>>(W1d, a1d, wd1, F_IN, H1);
  long long waves = (N + 15) / 16;
  k_gemm1_wmma<<<nb(waves * 32, B), B, 0, stream>>>(xb, Wp, hs, N);
  k_alpha1<<<nb(N, B), B, 0, stream>>>(hs, x, a1s, wd1, as1, ad1, N);

  // ---- layer 1 edges (segment softmax + aggregate) ----
  k_fill_u32<<<nb(N, B), B, 0, stream>>>(m1, 0u, (long long)N);
  k_fill_f32<<<nb(N, B), B, 0, stream>>>(s1, 0.f, (long long)N);
  k_fill_f32<<<nb((long long)N * H1, B), B, 0, stream>>>(out1, 0.f, (long long)N * H1);
  k_fill_f32<<<1, 128, 0, stream>>>(bnsum, 0.f, 128);
  k_edge_score<<<nb(T, B), B, 0, stream>>>(esrc, edst, E, N, as1, ad1, e1, m1);
  k_edge_expsum<<<nb(T, B), B, 0, stream>>>(esrc, edst, E, N, e1, m1, s1);
  k_edge_aggr64<<<nb((long long)T * 8, B), B, 0, stream>>>(esrc, edst, E, N, e1, s1, hs, out1);

  // ---- batch norm + relu (in place on out1) ----
  k_bn_stats<<<256, 256, 0, stream>>>(out1, b1, bnsum, bnsq, N);
  k_bn_final<<<1, 64, 0, stream>>>(bnsum, bnsq, gam, bet, bnscale, bnshift, N);
  k_bn_apply<<<nb((long long)N * H1, B), B, 0, stream>>>(out1, b1, bnscale, bnshift, (long long)N * H1);

  // ---- layer 2 ----
  k_wdvec<<<1, 64, 0, stream>>>(W2d, a2d, wd2, 64, 2);
  k_proj2<<<nb(N, B), B, 0, stream>>>(out1, W2s, a2s, wd2, hs2, as2, ad2, N);
  k_fill_u32<<<nb(N, B), B, 0, stream>>>(m2, 0u, (long long)N);
  k_fill_f32<<<nb(N, B), B, 0, stream>>>(s2, 0.f, (long long)N);
  k_init_out2<<<nb((long long)2 * N, B), B, 0, stream>>>((float*)d_out, b2, N);
  k_edge_score<<<nb(T, B), B, 0, stream>>>(esrc, edst, E, N, as2, ad2, e2, m2);
  k_edge_expsum<<<nb(T, B), B, 0, stream>>>(esrc, edst, E, N, e2, m2, s2);
  k_edge_aggr2<<<nb(T, B), B, 0, stream>>>(esrc, edst, E, N, e2, s2, hs2, (float*)d_out);
}
